// Sender_30150670418386
// MI455X (gfx1250) — matrix operand order repaired and checked by reading, not compiled, not run on hardware
//
#include <hip/hip_runtime.h>
#include <hip/hip_bf16.h>
#include <math.h>

// ---------------------------------------------------------------------------
// GATv2 (in_channels=1) forward on MI455X / gfx1250.
// Rank-1 structure of all linear layers collapses message aggregation to a
// per-(node,head) scalar S; the final FC is the only real GEMM and runs on
// v_wmma_f32_16x16x32_f16 with LDS-staged B and register-reused A.
// ---------------------------------------------------------------------------

typedef __attribute__((ext_vector_type(16))) _Float16 v16h;
typedef __attribute__((ext_vector_type(8)))  float    v8f;

#define NEG_SLOPE 0.2f
#define HE   128   // H * EMB
#define EMBC 64
#define HEADS 2
#define HID  128
#define BSTRIDE 136  // padded LDS stride (halfs) -> bank-conflict-free reads

// float atomic max via signed/unsigned integer atomics (init to -inf bits)
__device__ __forceinline__ void atomicMaxF(float* addr, float val) {
    if (val >= 0.0f) atomicMax((int*)addr, __float_as_int(val));
    else             atomicMin((unsigned int*)addr, __float_as_uint(val));
}

// K0: zero/-inf init of accumulators (workspace is poisoned by harness)
__global__ void k_init(float* deg, float* amean, float* mmax, float* denom,
                       float* S, int N) {
    int i = blockIdx.x * blockDim.x + threadIdx.x;
    if (i >= N) return;
    deg[i]   = 0.0f;
    amean[i] = 0.0f;
    const float ninf = __int_as_float(0xFF800000);
    mmax[2 * i + 0] = ninf;  mmax[2 * i + 1] = ninf;
    denom[2 * i + 0] = 0.0f; denom[2 * i + 1] = 0.0f;
    S[2 * i + 0] = 0.0f;     S[2 * i + 1] = 0.0f;
}

// K1: in-degree and incoming edge_attr sum (for fill_value='mean' self loops)
__global__ void k_deg(const int* __restrict__ dst, const float* __restrict__ ea,
                      float* deg, float* asum, int E) {
    int e = blockIdx.x * blockDim.x + threadIdx.x;
    if (e >= E) return;
    int d = dst[e];
    atomicAdd(&deg[d], 1.0f);
    atomicAdd(&asum[d], ea[e]);
}

// K2: attr_mean = asum / max(deg,1)  (in place)
__global__ void k_mean(float* amean, const float* __restrict__ deg, int N) {
    int i = blockIdx.x * blockDim.x + threadIdx.x;
    if (i >= N) return;
    amean[i] = amean[i] / fmaxf(deg[i], 1.0f);
}

// K3: per-edge GATv2 logits (128-wide LeakyReLU dot, weights staged in LDS)
//     + segment max via float atomic max. Edges [0,E) real, [E,E+N) self loops.
__global__ __launch_bounds__(256)
void k_logits(const float* __restrict__ x, const int* __restrict__ src,
              const int* __restrict__ dst, const float* __restrict__ eattr,
              const float* __restrict__ amean,
              const float* __restrict__ Wl, const float* __restrict__ bl,
              const float* __restrict__ Wr, const float* __restrict__ br,
              const float* __restrict__ We, const float* __restrict__ attv,
              float* __restrict__ logits, float* mmax, int E, int N) {
    __shared__ float sWl[HE], sWr[HE], sWe[HE], sAt[HE], sBs[HE];
    if (threadIdx.x < HE) {
        int t = threadIdx.x;
        sWl[t] = Wl[t]; sWr[t] = Wr[t]; sWe[t] = We[t];
        sAt[t] = attv[t]; sBs[t] = bl[t] + br[t];
    }
    __syncthreads();
    int e = blockIdx.x * blockDim.x + threadIdx.x;
    if (e >= E + N) return;
    float xs, xd, ea; int d;
    if (e < E) {
        int s = src[e]; d = dst[e];
        xs = x[s]; xd = x[d]; ea = eattr[e];
    } else {
        int n = e - E; d = n;
        xs = x[n]; xd = xs; ea = amean[n];
    }
#pragma unroll
    for (int h = 0; h < HEADS; ++h) {
        float acc = 0.0f;
#pragma unroll 8
        for (int c = 0; c < EMBC; ++c) {
            int k = h * EMBC + c;
            float arg = fmaf(xs, sWl[k], fmaf(xd, sWr[k], fmaf(ea, sWe[k], sBs[k])));
            float lr  = arg > 0.0f ? arg : NEG_SLOPE * arg;
            acc = fmaf(sAt[k], lr, acc);
        }
        logits[2 * e + h] = acc;
        atomicMaxF(&mmax[2 * d + h], acc);
    }
}

// K4: expv = exp(logit - max) (logits overwritten in place), denom accumulate
__global__ void k_exp(float* logits, const int* __restrict__ dst,
                      const float* __restrict__ mmax, float* denom, int E, int N) {
    int e = blockIdx.x * blockDim.x + threadIdx.x;
    if (e >= E + N) return;
    int d = (e < E) ? dst[e] : (e - E);
#pragma unroll
    for (int h = 0; h < HEADS; ++h) {
        float v = __expf(logits[2 * e + h] - mmax[2 * d + h]);
        logits[2 * e + h] = v;
        atomicAdd(&denom[2 * d + h], v);
    }
}

// K5: S[d,h] += alpha * x[src]  (the entire message aggregation, rank-1 trick)
__global__ void k_scatter(const float* __restrict__ logits,
                          const int* __restrict__ src, const int* __restrict__ dst,
                          const float* __restrict__ x, const float* __restrict__ denom,
                          float* S, int E, int N) {
    int e = blockIdx.x * blockDim.x + threadIdx.x;
    if (e >= E + N) return;
    int s, d;
    if (e < E) { s = src[e]; d = dst[e]; } else { s = d = e - E; }
    float xs = x[s];
#pragma unroll
    for (int h = 0; h < HEADS; ++h) {
        float alpha = logits[2 * e + h] / denom[2 * d + h];
        atomicAdd(&S[2 * d + h], alpha * xs);
    }
}

// K6: graph_emb (f16 for WMMA): emb[n,k] = Wl[k]*S[n,h(k)] + bl[k] + bias_out[k]
__global__ void k_emb(const float* __restrict__ S, const float* __restrict__ Wl,
                      const float* __restrict__ bl, const float* __restrict__ bo,
                      _Float16* __restrict__ emb, int N) {
    int i = blockIdx.x * blockDim.x + threadIdx.x;
    if (i >= N * HE) return;
    int n = i >> 7, k = i & (HE - 1), h = k >> 6;
    emb[i] = (_Float16)(fmaf(Wl[k], S[2 * n + h], bl[k] + bo[k]));
}

// K7: fold target-node half of `combined` + b_fc into a per-column constant
__global__ __launch_bounds__(HE)
void k_cvec(const float* __restrict__ S, const float* __restrict__ Wl,
            const float* __restrict__ bl, const float* __restrict__ bo,
            const float* __restrict__ Wfc, const float* __restrict__ bfc,
            const int* __restrict__ tidx, float* __restrict__ cvec) {
    __shared__ float tgt[HE];
    int t = threadIdx.x;
    int tn = tidx[0];
    int h = t >> 6;
    tgt[t] = fmaf(Wl[t], S[2 * tn + h], bl[t] + bo[t]);
    __syncthreads();
    float acc = bfc[t];
#pragma unroll 8
    for (int k = 0; k < HE; ++k)
        acc = fmaf(Wfc[t * (2 * HE) + HE + k], tgt[k], acc);
    cvec[t] = acc;
}

// K8: left half of W_fc -> f16, [j][k] row-major (B-operand friendly)
__global__ void k_wfc16(const float* __restrict__ Wfc, _Float16* __restrict__ w16,
                        int count) {
    int i = blockIdx.x * blockDim.x + threadIdx.x;
    if (i >= count) return;
    int j = i >> 7, k = i & (HE - 1);
    w16[i] = (_Float16)Wfc[j * (2 * HE) + k];
}

// K9: out[N,128] = emb[N,128] @ w16^T + cvec  via v_wmma_f32_16x16x32_f16.
//     256 threads = 8 waves; each wave owns a DISTINCT 16-row strip and
//     computes all 8 column tiles (8x v8f accumulators), so A is loaded from
//     global exactly once and reused across 8 WMMAs per K-chunk. B (32KB) is
//     staged once per block into LDS with padded stride (ds_load_b128 reads,
//     bank = 4*lane mod 64 -> conflict-free within a half-wave).
//     A lane layout: lane(m,hi) holds row m, K in [K0+8hi,+8) u [K0+16+8hi,+8).
//     B lane layout: column n=lane&15, K in [K0+8hi,+8) u [K0+16+8hi,+8).
//     D: VGPR r -> row (hi*8 + r), col n.
__global__ __launch_bounds__(256)
void k_fc_wmma(const _Float16* __restrict__ emb, const _Float16* __restrict__ w16,
               const float* __restrict__ cvec, float* __restrict__ out, int N) {
    __shared__ _Float16 sB[HID * BSTRIDE];  // [col][k], padded

    // ---- stage B: 128 cols x 128 halfs, 16B chunks, all 256 threads ----
#pragma unroll
    for (int c = threadIdx.x; c < HID * 16; c += 256) {
        int col = c >> 4, koff = (c & 15) * 8;
        *reinterpret_cast<float4*>(&sB[col * BSTRIDE + koff]) =
            *reinterpret_cast<const float4*>(w16 + col * HE + koff);
    }
    __syncthreads();

    const int wave = threadIdx.x >> 5;
    const int lane = threadIdx.x & 31;
    const int m    = lane & 15;
    const int hi   = lane >> 4;
    const int row0 = (blockIdx.x * 8 + wave) * 16;
    if (row0 >= N) return;

    const _Float16* arow = emb + (size_t)(row0 + m) * HE;

    v8f acc[8];
#pragma unroll
    for (int ct = 0; ct < 8; ++ct) acc[ct] = (v8f){};

#pragma unroll
    for (int kc = 0; kc < 4; ++kc) {
        const int K0 = kc * 32;
        v16h a;
        reinterpret_cast<float4*>(&a)[0] =
            *reinterpret_cast<const float4*>(arow + K0 + hi * 8);
        reinterpret_cast<float4*>(&a)[1] =
            *reinterpret_cast<const float4*>(arow + K0 + 16 + hi * 8);
#pragma unroll
        for (int ct = 0; ct < 8; ++ct) {
            v16h b;
            const float4* pb = reinterpret_cast<const float4*>(
                &sB[(ct * 16 + m) * BSTRIDE + K0 + hi * 8]);
            reinterpret_cast<float4*>(&b)[0] = pb[0];
            reinterpret_cast<float4*>(&b)[1] = pb[2];
            // (neg_a, A, neg_b, B, c_mod, C, reuse_a, reuse_b)
            acc[ct] = __builtin_amdgcn_wmma_f32_16x16x32_f16(
                false, a, false, b, (short)0, acc[ct], false, false);
        }
    }

    if (row0 + 16 <= N) {  // uniform fast path: straight-line stores
#pragma unroll
        for (int ct = 0; ct < 8; ++ct) {
            const int   col  = ct * 16 + m;
            const float cadd = cvec[col];
            float* orow = out + (size_t)(row0 + hi * 8) * HID + col;
#pragma unroll
            for (int r = 0; r < 8; ++r) orow[(size_t)r * HID] = acc[ct][r] + cadd;
        }
    } else {               // ragged tail (unused when N % 16 == 0)
#pragma unroll
        for (int ct = 0; ct < 8; ++ct) {
            const int   col  = ct * 16 + m;
            const float cadd = cvec[col];
#pragma unroll
            for (int r = 0; r < 8; ++r) {
                int rowD = row0 + hi * 8 + r;
                if (rowD < N) out[(size_t)rowD * HID + col] = acc[ct][r] + cadd;
            }
        }
    }
}

extern "C" void kernel_launch(void* const* d_in, const int* in_sizes, int n_in,
                              void* d_out, int out_size, void* d_ws, size_t ws_size,
                              hipStream_t stream) {
    const int N = in_sizes[0];          // nodes
    const int E = in_sizes[2];          // edges
    const int T = E + N;                // edges incl. self loops

    const float* x     = (const float*)d_in[0];
    const int*   eidx  = (const int*)  d_in[1];
    const float* eattr = (const float*)d_in[2];
    const int*   tidx  = (const int*)  d_in[3];
    const float* Wl    = (const float*)d_in[4];
    const float* bl    = (const float*)d_in[5];
    const float* Wr    = (const float*)d_in[6];
    const float* br    = (const float*)d_in[7];
    const float* We    = (const float*)d_in[8];
    const float* attv  = (const float*)d_in[9];
    const float* bo    = (const float*)d_in[10];
    const float* Wfc   = (const float*)d_in[11];
    const float* bfc   = (const float*)d_in[12];
    float* out = (float*)d_out;

    const int* src = eidx;          // edge_index[0]
    const int* dst = eidx + E;      // edge_index[1]

    // ---- workspace carve-up (floats first, then f16, 16B-aligned) ----
    float* ws = (float*)d_ws;
    float* deg    = ws;                 ws += N;
    float* amean  = ws;                 ws += N;       // asum -> mean in place
    float* mmax   = ws;                 ws += 2 * (size_t)N;
    float* denom  = ws;                 ws += 2 * (size_t)N;
    float* S      = ws;                 ws += 2 * (size_t)N;
    float* logits = ws;                 ws += 2 * (size_t)T;
    float* cvec   = ws;                 ws += HE;
    size_t off = (size_t)((char*)ws - (char*)d_ws);
    off = (off + 15) & ~(size_t)15;
    _Float16* emb = (_Float16*)((char*)d_ws + off);
    _Float16* w16 = emb + (size_t)N * HE;
    (void)ws_size; (void)n_in; (void)out_size;

    const int B = 256;
    k_init   <<<(N + B - 1) / B, B, 0, stream>>>(deg, amean, mmax, denom, S, N);
    k_deg    <<<(E + B - 1) / B, B, 0, stream>>>(dst, eattr, deg, amean, E);
    k_mean   <<<(N + B - 1) / B, B, 0, stream>>>(amean, deg, N);
    k_logits <<<(T + B - 1) / B, B, 0, stream>>>(x, src, dst, eattr, amean,
                                                 Wl, bl, Wr, br, We, attv,
                                                 logits, mmax, E, N);
    k_exp    <<<(T + B - 1) / B, B, 0, stream>>>(logits, dst, mmax, denom, E, N);
    k_scatter<<<(T + B - 1) / B, B, 0, stream>>>(logits, src, dst, x, denom, S, E, N);
    k_emb    <<<((size_t)N * HE + B - 1) / B, B, 0, stream>>>(S, Wl, bl, bo, emb, N);
    k_cvec   <<<1, HE, 0, stream>>>(S, Wl, bl, bo, Wfc, bfc, tidx, cvec);
    k_wfc16  <<<(HID * HE + B - 1) / B, B, 0, stream>>>(Wfc, w16, HID * HE);

    const int nwaves  = (N + 15) / 16;            // one wave per 16-row strip
    const int nblocks = (nwaves + 7) / 8;         // 8 waves per block
    k_fc_wmma<<<nblocks, 256, 0, stream>>>(emb, w16, cvec, out, N);
}
